// SpeechEnhancementModel_8083128451708
// MI455X (gfx1250) — matrix operand order repaired and checked by reading, not compiled, not run on hardware
//
#include <hip/hip_runtime.h>
#include <math.h>

// MI455X / gfx1250 (CDNA5, wave32). GEMM-shaped work -> v_wmma_f32_16x16x32_f16
// (fp16 operands, fp32 accumulate). GEMM operands are mirrored in fp16 in global
// memory so tile staging uses GLOBAL_LOAD_ASYNC_TO_LDS_B128 (ASYNCcnt) straight
// into fragment-ordered LDS; fragment reads are contiguous ds_load_b128s.
typedef __attribute__((ext_vector_type(16))) _Float16 v16h;
typedef __attribute__((ext_vector_type(8)))  _Float16 v8h;
typedef __attribute__((ext_vector_type(8)))  float    v8f;

#define CDIV(a,b) (((a)+(b)-1)/(b))

__device__ __forceinline__ float d_sigmoid(float x){ return 1.f/(1.f+expf(-x)); }
__device__ __forceinline__ float d_softplus(float x){ return (x > 20.f) ? x : log1pf(expf(x)); }

__device__ __forceinline__ v8h pack8(float4 a, float4 b){
  v8h h;
  h[0]=(_Float16)a.x; h[1]=(_Float16)a.y; h[2]=(_Float16)a.z; h[3]=(_Float16)a.w;
  h[4]=(_Float16)b.x; h[5]=(_Float16)b.y; h[6]=(_Float16)b.z; h[7]=(_Float16)b.w;
  return h;
}
__device__ __forceinline__ unsigned lds_addr32(const void* p){
  return (unsigned)(size_t)p;     // flat shared ptr: LDS offset lives in addr[31:0]
}

// ===================== fp32 -> fp16 mirror conversion =================
__global__ __launch_bounds__(256) void k_f32_to_f16(
    const float* __restrict__ s, _Float16* __restrict__ d, long n)
{
  long gid = (long)blockIdx.x*256 + threadIdx.x;
  if (gid < n) d[gid] = (_Float16)s[gid];
}

// ============================ encoder stem ============================
__global__ __launch_bounds__(256) void k_conv1x1(
    const float* __restrict__ amp, const float* __restrict__ pha,
    const float* __restrict__ w1, const float* __restrict__ b1,
    float* __restrict__ Y, int Bn, int Tn, int Ff)
{
  long n = (long)Bn*64*Tn*Ff;
  long gid = (long)blockIdx.x*256 + threadIdx.x;
  if (gid >= n) return;
  int f = (int)(gid % Ff); long r = gid / Ff;
  int t = (int)(r % Tn); r /= Tn;
  int c = (int)(r % 64); int b = (int)(r / 64);
  long ia = ((long)b*Ff + f)*Tn + t;                 // inputs are (B,F,T)
  Y[gid] = w1[c*2+0]*amp[ia] + w1[c*2+1]*pha[ia] + b1[c];
}

// ======================= instance norm + PReLU ========================
__global__ __launch_bounds__(256) void k_inorm_prelu(
    const float* __restrict__ X, int inC,
    float* __restrict__ Y, int outC, int outCoff,
    const float* __restrict__ g, const float* __restrict__ bt, const float* __restrict__ a,
    int plane, int C)
{
  int bc = blockIdx.x;
  int b = bc / C, c = bc % C;
  const float* xp = X + ((long)b*inC + c)*(long)plane;
  float* yp = Y + ((long)b*outC + outCoff + c)*(long)plane;
  __shared__ float s1[256], s2[256];
  float sum = 0.f, sq = 0.f;
  for (int i = threadIdx.x; i < plane; i += 256) { float v = xp[i]; sum += v; sq += v*v; }
  s1[threadIdx.x] = sum; s2[threadIdx.x] = sq;
  __syncthreads();
  for (int off = 128; off; off >>= 1) {
    if (threadIdx.x < (unsigned)off) { s1[threadIdx.x] += s1[threadIdx.x+off]; s2[threadIdx.x] += s2[threadIdx.x+off]; }
    __syncthreads();
  }
  float mean = s1[0] / (float)plane;
  float var  = s2[0] / (float)plane - mean*mean;
  float inv  = rsqrtf(var + 1e-5f);
  float gg = g[c]*inv;
  float bb = bt[c] - mean*gg;
  float aa = a[c];
  for (int i = threadIdx.x; i < plane; i += 256) {
    float v = xp[i]*gg + bb;
    yp[i] = (v >= 0.f) ? v : aa*v;
  }
}

// ===================== generic WMMA GEMM (A16 * W16^T) ================
// C[m, coff+n] = act( sum_k A[m,k]*W[n,k] + bias[n] ) + res[m,n]
// A, W are fp16 mirrors. Staging: one global_load_async_to_lds_b128 per thread
// per K-step directly into fragment-ordered LDS (Asf[wm][lane][16]).
__global__ __launch_bounds__(256) void k_gemm_wmma(
    const _Float16* __restrict__ A, int lda,
    const _Float16* __restrict__ W,
    float* __restrict__ C, int ldc, int coff,
    int M, int N, int K,
    const float* __restrict__ bias,
    const float* __restrict__ res, int ldres,
    int act)
{
  __shared__ _Float16 Asf[4][32][16];   // 4 KB, fragment order
  __shared__ _Float16 Bsf[4][32][16];   // 4 KB, fragment order
  int mtile = blockIdx.x*64, ntile = blockIdx.y*64;
  int tid = threadIdx.x;
  int lane = tid & 31, wave = tid >> 5;
  int wm = wave >> 1, wn = wave & 1;
  int r = lane & 15, hf = lane >> 4;

  // staging decomposition: thread -> (sg, sl, sp), 8 contiguous k (16 bytes)
  int sg = tid >> 6;            // A: wm-subtile / B: 16-col group
  int sl = (tid >> 1) & 31;     // fragment lane
  int sp = tid & 1;             // half of the 16 fragment elements
  int srr = sl & 15, shh = sl >> 4;

  int m_st = mtile + sg*16 + srr;
  bool mval = m_st < M;
  const _Float16* agp = A + (long)m_st*lda + shh*8 + sp*16;  // A frag: k = h*8+j (+8 for j>=8)
  int n_st = ntile + sg*16 + srr;
  bool nval = n_st < N;
  const _Float16* wgp = W + (long)n_st*K + shh*8 + sp*16;    // B frag: k = h*8+j (+8 for j>=8)

  unsigned ldsA = lds_addr32(&Asf[sg][sl][sp*8]);
  unsigned ldsB = lds_addr32(&Bsf[sg][sl][sp*8]);
  if (!mval) { v8h z = {}; *(v8h*)&Asf[sg][sl][sp*8] = z; }  // zero tails once
  if (!nval) { v8h z = {}; *(v8h*)&Bsf[sg][sl][sp*8] = z; }

  v8f acc0 = {}; v8f acc1 = {};
  for (int k0 = 0; k0 < K; k0 += 32) {
    if (mval)
      asm volatile("global_load_async_to_lds_b128 %0, %1, off"
                   :: "v"(ldsA), "v"(agp + k0) : "memory");
    if (nval)
      asm volatile("global_load_async_to_lds_b128 %0, %1, off"
                   :: "v"(ldsB), "v"(wgp + k0) : "memory");
    asm volatile("s_wait_asynccnt 0x0" ::: "memory");
    __syncthreads();
    v16h af  = *(const v16h*)&Asf[wm][lane][0];
    v16h bf0 = *(const v16h*)&Bsf[wn*2+0][lane][0];
    v16h bf1 = *(const v16h*)&Bsf[wn*2+1][lane][0];
    acc0 = __builtin_amdgcn_wmma_f32_16x16x32_f16(false, af, false, bf0, (short)0, acc0, false, false);
    acc1 = __builtin_amdgcn_wmma_f32_16x16x32_f16(false, af, false, bf1, (short)0, acc1, false, false);
    __syncthreads();
  }
#pragma unroll
  for (int v = 0; v < 8; ++v) {
    int m = mtile + wm*16 + hf*8 + v;
    if (m >= M) continue;
#pragma unroll
    for (int piece = 0; piece < 2; ++piece) {
      int n = ntile + wn*32 + piece*16 + r;
      if (n >= N) continue;
      float val = piece ? acc1[v] : acc0[v];
      if (bias) val += bias[n];
      if (act == 1) val = d_softplus(val);
      if (res) val += res[(long)m*ldres + n];
      C[(long)m*ldc + coff + n] = val;
    }
  }
}

// ================= implicit-GEMM conv (WMMA, im2col in LDS) ==========
// X: (B, Ctot, Tn, Fin) fp32, channel window at Cbase. W: (Cout,Cin,KT,3) fp32.
// kernel (KT,3), time dilation dil (causal), freq stride with pad 1 each side.
// Optional r=2 pixel-shuffle write.
__global__ __launch_bounds__(256) void k_conv_wmma(
    const float* __restrict__ X, int Ctot, int Cbase,
    const float* __restrict__ W, const float* __restrict__ bias,
    float* __restrict__ Y,
    int Bn, int Tn, int Fin, int Fout,
    int Cin, int Cout, int KT, int dil, int stride, int shuffle)
{
  const int M = Bn * Tn * Fout;
  const int K = Cin * KT * 3;
  __shared__ _Float16 Asf[4][32][16];
  __shared__ _Float16 Bsf[4][32][16];
  int mtile = blockIdx.x*64, ntile = blockIdx.y*64;
  int tid = threadIdx.x;
  int lane = tid & 31, wave = tid >> 5;
  int wm = wave >> 1, wn = wave & 1;
  int r = lane & 15, hf = lane >> 4;

  int sg = tid >> 6;
  int sl = (tid >> 1) & 31;
  int sp = tid & 1;
  int srr = sl & 15, shh = sl >> 4;

  int m_st = mtile + sg*16 + srr;
  bool mval = m_st < M;
  int b_st = 0, t_st = 0, fo_st = 0;
  if (mval) {
    b_st = m_st / (Tn*Fout); int mrem = m_st - b_st*(Tn*Fout);
    t_st = mrem / Fout;      fo_st = mrem - t_st*Fout;
  }
  const int kboff = shh*8 + sp*16;

  int n_st = ntile + sg*16 + srr;
  bool nval = n_st < Cout;
  const float* wrow = W + (long)n_st*K + shh*8 + sp*16;

  v8f acc0 = {}; v8f acc1 = {};
  for (int k0 = 0; k0 < K; k0 += 32) {
    v8h ha = {};
    if (mval) {
      int k = k0 + kboff;
      int ic = k / (KT*3); int rem = k - ic*(KT*3);
      int kt = rem / 3;    int kf  = rem - kt*3;
#pragma unroll
      for (int j = 0; j < 8; ++j) {
        int tin = t_st + (kt - (KT-1))*dil;
        int fin = fo_st*stride + kf - 1;
        float v = 0.f;
        if (tin >= 0 && fin >= 0 && fin < Fin)
          v = X[(((long)b_st*Ctot + Cbase + ic)*Tn + tin)*Fin + fin];
        ha[j] = (_Float16)v;
        if (++kf == 3) { kf = 0; if (++kt == KT) { kt = 0; ++ic; } }
      }
    }
    *(v8h*)&Asf[sg][sl][sp*8] = ha;
    float4 b0 = make_float4(0.f,0.f,0.f,0.f), b1 = b0;
    if (nval) { b0 = *(const float4*)(wrow + k0); b1 = *(const float4*)(wrow + k0 + 4); }
    *(v8h*)&Bsf[sg][sl][sp*8] = pack8(b0, b1);
    __syncthreads();
    v16h af  = *(const v16h*)&Asf[wm][lane][0];
    v16h bf0 = *(const v16h*)&Bsf[wn*2+0][lane][0];
    v16h bf1 = *(const v16h*)&Bsf[wn*2+1][lane][0];
    acc0 = __builtin_amdgcn_wmma_f32_16x16x32_f16(false, af, false, bf0, (short)0, acc0, false, false);
    acc1 = __builtin_amdgcn_wmma_f32_16x16x32_f16(false, af, false, bf1, (short)0, acc1, false, false);
    __syncthreads();
  }
#pragma unroll
  for (int v = 0; v < 8; ++v) {
    int m = mtile + wm*16 + hf*8 + v;
    if (m >= M) continue;
    int b = m / (Tn*Fout); int mrem = m - b*(Tn*Fout);
    int t = mrem / Fout;   int fo = mrem - t*Fout;
#pragma unroll
    for (int piece = 0; piece < 2; ++piece) {
      int n = ntile + wn*32 + piece*16 + r;
      if (n >= Cout) continue;
      float val = (piece ? acc1[v] : acc0[v]) + (bias ? bias[n] : 0.f);
      long off;
      if (!shuffle) off = (((long)b*Cout + n)*Tn + t)*Fout + fo;
      else { int c = n & 63; int rr = n >> 6;
             off = (((long)b*64 + c)*Tn + t)*(long)(2*Fout) + (long)fo*2 + rr; }
      Y[off] = val;
    }
  }
}

// ============ fused transpose + LayerNorm over C=64 (two mappings) ====
// Writes fp32 (for residuals) and an fp16 mirror (GEMM A operand).
__global__ __launch_bounds__(256) void k_ln_rows(
    const float* __restrict__ X, float* __restrict__ Y, _Float16* __restrict__ Y16,
    const float* __restrict__ w, const float* __restrict__ b,
    int Bn, int Tn, int F2, int mode)
{
  __shared__ float s1[256], s2[256];
  int gr = threadIdx.x >> 6, c = threadIdx.x & 63;
  long row = (long)blockIdx.x*4 + gr;
  long Mtot = (long)Bn*Tn*F2;
  bool valid = row < Mtot;
  float v = 0.f;
  if (valid) {
    if (mode == 0) {               // row = (b*F2+f)*T + t   from x (B,64,T,F2)
      long s = row / Tn; int t = (int)(row % Tn);
      int bb = (int)(s / F2), f = (int)(s % F2);
      v = X[(((long)bb*64 + c)*Tn + t)*F2 + f];
    } else {                        // row = (b*T+t)*F2 + f  from rows (b*F2+f)*T+t
      long bt_ = row / F2; int f = (int)(row % F2);
      int bb = (int)(bt_ / Tn), t = (int)(bt_ % Tn);
      v = X[(((long)bb*F2 + f)*Tn + t)*64 + c];
    }
  }
  s1[threadIdx.x] = v; s2[threadIdx.x] = v*v;
  __syncthreads();
  for (int off = 32; off; off >>= 1) {
    if (c < off) { s1[threadIdx.x] += s1[threadIdx.x+off]; s2[threadIdx.x] += s2[threadIdx.x+off]; }
    __syncthreads();
  }
  int base = gr*64;
  float mean = s1[base] * (1.f/64.f);
  float var  = s2[base] * (1.f/64.f) - mean*mean;
  float inv  = rsqrtf(var + 1e-5f);
  if (valid) {
    float o = (v - mean)*inv*w[c] + b[c];
    Y[row*64 + c] = o;
    Y16[row*64 + c] = (_Float16)o;
  }
}

// ============================== xLSTM scan ============================
__global__ __launch_bounds__(256) void k_xlstm_scan(
    const float* __restrict__ G, _Float16* __restrict__ H16, int nseq, int Tn)
{
  long gid = (long)blockIdx.x*256 + threadIdx.x;
  if (gid >= (long)nseq*64) return;
  long s = gid >> 6; int c = (int)(gid & 63);
  float cn = 0.f, cs = 0.f;
  const float* gp = G + s*(long)Tn*256 + c;
  _Float16* hp = H16 + s*(long)Tn*64 + c;
  for (int t = 0; t < Tn; ++t) {
    float o  = d_sigmoid(gp[0]);
    float iv = expf(gp[64]);
    float fe = expf(d_sigmoid(gp[128]));
    float gv = tanhf(gp[192]);
    cn = fe*cn + iv*gv;
    cs = fe*cs + iv;
    hp[0] = (_Float16)(o*(cn/cs));
    gp += 256; hp += 64;
  }
}

// =========== mamba depthwise causal conv(4) + SiLU on xm half ========
__global__ __launch_bounds__(256) void k_dwconv_silu(
    const float* __restrict__ XR, const float* __restrict__ cw, const float* __restrict__ cb,
    float* __restrict__ XMA, _Float16* __restrict__ XMA16, int L, long Mtot)
{
  long gid = (long)blockIdx.x*256 + threadIdx.x;
  if (gid >= Mtot*128) return;
  long m = gid >> 7; int ch = (int)(gid & 127);
  long s = m / L; int l = (int)(m % L);
  float acc = cb[ch];
#pragma unroll
  for (int k = 0; k < 4; ++k) {
    int li = l + k - 3;
    if (li >= 0) acc += cw[ch*4 + k] * XR[((s*(long)L + li)*256) + ch];
  }
  float y = acc * d_sigmoid(acc);           // silu
  XMA[m*128 + ch] = y;
  XMA16[m*128 + ch] = (_Float16)y;
}

// ======================== mamba selective scan =======================
__global__ __launch_bounds__(256) void k_mamba_scan(
    const float* __restrict__ Dl, const float* __restrict__ BC,
    const float* __restrict__ XMA, const float* __restrict__ XR,
    const float* __restrict__ A_log, const float* __restrict__ Dp,
    _Float16* __restrict__ Y16, int nseq, int L)
{
  long gid = (long)blockIdx.x*256 + threadIdx.x;
  if (gid >= (long)nseq*128) return;
  long s = gid >> 7; int ch = (int)(gid & 127);
  float Aj[16], h[16];
#pragma unroll
  for (int j = 0; j < 16; ++j) { Aj[j] = -expf(A_log[ch*16 + j]); h[j] = 0.f; }
  float Dv = Dp[ch];
  for (int l = 0; l < L; ++l) {
    long m = s*(long)L + l;
    float d  = Dl[m*128 + ch];
    float xt = XMA[m*128 + ch];
    float dx = d*xt;
    float y = 0.f;
#pragma unroll
    for (int j = 0; j < 16; ++j) {
      h[j] = expf(d*Aj[j])*h[j] + dx*BC[m*32 + j];
      y += h[j]*BC[m*32 + 16 + j];
    }
    y += xt*Dv;
    float rr = XR[m*256 + 128 + ch];
    y *= rr * d_sigmoid(rr);                // * silu(res)
    Y16[m*128 + ch] = (_Float16)y;
  }
}

// =================== un-transpose + block residual add ===============
__global__ __launch_bounds__(256) void k_untranspose_add(
    const float* __restrict__ XF, float* __restrict__ X, int Bn, int Tn, int F2)
{
  long n = (long)Bn*64*Tn*F2;
  long gid = (long)blockIdx.x*256 + threadIdx.x;
  if (gid >= n) return;
  int f = (int)(gid % F2); long r = gid / F2;
  int t = (int)(r % Tn); r /= Tn;
  int c = (int)(r % 64); int b = (int)(r / 64);
  X[gid] += XF[(((long)b*Tn + t)*F2 + f)*64 + c];
}

// ============== copy x into a 320-ch skip buffer at offset 256 =======
__global__ __launch_bounds__(256) void k_copy_channels(
    const float* __restrict__ X, float* __restrict__ Y, long n, long plane)
{
  long gid = (long)blockIdx.x*256 + threadIdx.x;
  if (gid >= n) return;
  long b = gid / (64*plane); long rem = gid - b*64*plane;
  long c = rem / plane; long i = rem - c*plane;
  Y[((b*320 + 256 + c)*plane) + i] = X[gid];
}

// ============== 1-channel output convs (kernel (1,2), left pad) ======
__global__ __launch_bounds__(256) void k_maskconv(
    const float* __restrict__ Mb, const float* __restrict__ ow, const float* __restrict__ ob,
    float* __restrict__ out, int Bn, int Tn, int Fo, int Wd)
{
  long n = (long)Bn*Fo*Tn;
  long gid = (long)blockIdx.x*256 + threadIdx.x;
  if (gid >= n) return;
  int t = (int)(gid % Tn); long r = gid / Tn;
  int f = (int)(r % Fo); int b = (int)(r / Fo);
  const float* base = Mb + (long)b*64*Tn*Wd;
  float acc = ob[0];
  for (int c = 0; c < 64; ++c) {
    long o = ((long)c*Tn + t)*Wd + f;
    float xm1 = (f >= 1) ? base[o-1] : 0.f;
    acc += ow[c*2+0]*xm1 + ow[c*2+1]*base[o];
  }
  out[gid] = d_sigmoid(acc);
}

__global__ __launch_bounds__(256) void k_phaseconv(
    const float* __restrict__ Mb,
    const float* __restrict__ rw, const float* __restrict__ rb,
    const float* __restrict__ iw, const float* __restrict__ ib,
    float* __restrict__ out, int Bn, int Tn, int Fo, int Wd)
{
  long n = (long)Bn*Fo*Tn;
  long gid = (long)blockIdx.x*256 + threadIdx.x;
  if (gid >= n) return;
  int t = (int)(gid % Tn); long r = gid / Tn;
  int f = (int)(r % Fo); int b = (int)(r / Fo);
  const float* base = Mb + (long)b*64*Tn*Wd;
  float ar = rb[0], ai = ib[0];
  for (int c = 0; c < 64; ++c) {
    long o = ((long)c*Tn + t)*Wd + f;
    float xm1 = (f >= 1) ? base[o-1] : 0.f;
    float x0  = base[o];
    ar += rw[c*2+0]*xm1 + rw[c*2+1]*x0;
    ai += iw[c*2+0]*xm1 + iw[c*2+1]*x0;
  }
  out[gid] = atan2f(ai, ar);
}

// =========================== output assembly =========================
__global__ __launch_bounds__(256) void k_final(
    const float* __restrict__ amp_in, const float* __restrict__ maskp,
    const float* __restrict__ phap, float* __restrict__ out, long NT)
{
  long gid = (long)blockIdx.x*256 + threadIdx.x;
  if (gid >= NT) return;
  float a = amp_in[gid]*maskp[gid];
  float p = phap[gid];
  out[gid] = a;
  out[NT + gid] = p;
  out[2*NT + gid*2 + 0] = a*cosf(p);
  out[2*NT + gid*2 + 1] = a*sinf(p);
}

// =====================================================================
extern "C" void kernel_launch(void* const* d_in, const int* in_sizes, int n_in,
                              void* d_out, int out_size, void* d_ws, size_t ws_size,
                              hipStream_t stream)
{
  (void)in_sizes; (void)n_in; (void)out_size; (void)ws_size;
  const int Bn=2, Tn=301, Ff=201, F2=101;
  const long planeF  = (long)Tn*Ff;     // 60501
  const long planeF2 = (long)Tn*F2;     // 30401
  const long plane202= (long)Tn*202;    // 60802
  const long M2 = (long)Bn*Tn*F2;       // 60802 rows for all block GEMMs
  const long NT = (long)Bn*Ff*Tn;       // 121002

  // ---- unpack inputs (JAX sorted-dict flattening order) ----
  int ii = 0;
  auto nx = [&]() { return (const float*)d_in[ii++]; };
  const float* amp_in = nx();
  const float* pha_in = nx();
  struct Blk { const float *ln1_b,*ln1_w,*ln2_b,*ln2_w,
    *A_log,*Dp,*b_w,*c_w,*conv_b,*conv_w,*dt_b,*dt_w,*in_w,*out_w,
    *w_f,*w_g,*w_i,*w_o,*w_out; } blk[4];
  for (int k=0;k<4;k++){ Blk& P=blk[k];
    P.ln1_b=nx(); P.ln1_w=nx(); P.ln2_b=nx(); P.ln2_w=nx();
    P.A_log=nx(); P.Dp=nx(); P.b_w=nx(); P.c_w=nx(); P.conv_b=nx(); P.conv_w=nx();
    P.dt_b=nx(); P.dt_w=nx(); P.in_w=nx(); P.out_w=nx();
    P.w_f=nx(); P.w_g=nx(); P.w_i=nx(); P.w_o=nx(); P.w_out=nx();
  }
  struct DenseL { const float *a,*b,*bt,*g,*w; };
  const float *e_a1=nx(), *e_a2=nx(), *e_b1=nx(), *e_b2=nx(), *e_bt1=nx(), *e_bt2=nx();
  DenseL ed[4]; for (int k=0;k<4;k++){ ed[k].a=nx(); ed[k].b=nx(); ed[k].bt=nx(); ed[k].g=nx(); ed[k].w=nx(); }
  const float *e_g1=nx(), *e_g2=nx(), *e_w1=nx(), *e_w2=nx();
  const float *m_a=nx(), *m_bt=nx();
  DenseL md[4]; for (int k=0;k<4;k++){ md[k].a=nx(); md[k].b=nx(); md[k].bt=nx(); md[k].g=nx(); md[k].w=nx(); }
  const float *m_g=nx(), *m_out_b=nx(), *m_out_w=nx(), *m_sp_b=nx(), *m_sp_w=nx();
  const float *p_a=nx(), *p_bt=nx();
  DenseL pd[4]; for (int k=0;k<4;k++){ pd[k].a=nx(); pd[k].b=nx(); pd[k].bt=nx(); pd[k].g=nx(); pd[k].w=nx(); }
  const float *p_g=nx(), *p_i_b=nx(), *p_i_w=nx(), *p_r_b=nx(), *p_r_w=nx(), *p_sp_b=nx(), *p_sp_w=nx();

  // ---- workspace bump allocator ----
  char* wsp = (char*)d_ws;
  auto alloc  = [&](long nfl) { float* p = (float*)wsp; wsp += ((nfl*4L + 255) & ~255L); return p; };
  auto alloch = [&](long nh)  { _Float16* p = (_Float16*)wsp; wsp += ((nh*2L + 255) & ~255L); return p; };
  float* convraw = alloc((long)Bn*64*planeF);    // raw conv output (reused everywhere)
  float* skipF   = alloc((long)Bn*320*planeF);   // encoder dense skip (prepend order reversed)
  float* x_cur   = alloc((long)Bn*64*planeF2);   // running (B,64,T,F2)
  float* xt_ln   = alloc(M2*64);
  float* gates   = alloc(M2*256);                // also reused as mamba xr
  float* xt_post = alloc(M2*64);
  float* xf_ln   = alloc(M2*64);
  float* xma     = alloc(M2*128);
  float* deltab  = alloc(M2*128);
  float* bc      = alloc(M2*32);
  float* xf_post = alloc(M2*64);
  float* skipF2  = alloc((long)Bn*320*planeF2);  // decoder dense skip (mask then phase)
  float* m202    = alloc((long)Bn*64*plane202);  // decoder tail (B,64,T,202)
  float* maskp   = alloc(NT);
  float* phap    = alloc(NT);
  // fp16 mirrors (GEMM A operands)
  _Float16* xt_ln16 = alloch(M2*64);
  _Float16* hbuf16  = alloch(M2*64);
  _Float16* xf_ln16 = alloch(M2*64);
  _Float16* xma16   = alloch(M2*128);
  _Float16* ymid16  = alloch(M2*128);
  // fp16 weight mirrors
  struct BW16 { _Float16 *o,*i,*f,*g,*out,*inw,*dtw,*bw,*cw,*ow; } bw[4];

  auto cvt = [&](const float* s, _Float16* d, long n){
    k_f32_to_f16<<<(unsigned)CDIV(n,256),256,0,stream>>>(s, d, n);
  };
  for (int kb=0;kb<4;kb++){
    const Blk& P = blk[kb];
    bw[kb].o   = alloch(64*64);   cvt(P.w_o,   bw[kb].o,   64*64);
    bw[kb].i   = alloch(64*64);   cvt(P.w_i,   bw[kb].i,   64*64);
    bw[kb].f   = alloch(64*64);   cvt(P.w_f,   bw[kb].f,   64*64);
    bw[kb].g   = alloch(64*64);   cvt(P.w_g,   bw[kb].g,   64*64);
    bw[kb].out = alloch(64*64);   cvt(P.w_out, bw[kb].out, 64*64);
    bw[kb].inw = alloch(256*64);  cvt(P.in_w,  bw[kb].inw, 256*64);
    bw[kb].dtw = alloch(128*128); cvt(P.dt_w,  bw[kb].dtw, 128*128);
    bw[kb].bw  = alloch(16*128);  cvt(P.b_w,   bw[kb].bw,  16*128);
    bw[kb].cw  = alloch(16*128);  cvt(P.c_w,   bw[kb].cw,  16*128);
    bw[kb].ow  = alloch(64*128);  cvt(P.out_w, bw[kb].ow,  64*128);
  }

  auto run_dense = [&](float* skip, const DenseL* dl, int Fx){
    long plane = (long)Tn*Fx;
    long Mloc = (long)Bn*plane;
    for (int il=0; il<4; ++il){
      int Cin = 64*(il+1), d = 1<<il, cbase = 256 - 64*il;
      dim3 gg((unsigned)CDIV(Mloc,64), 1);
      k_conv_wmma<<<gg,256,0,stream>>>(skip, 320, cbase, dl[il].w, dl[il].b, convraw,
          Bn, Tn, Fx, Fx, Cin, 64, /*KT=*/2, /*dil=*/d, /*stride=*/1, /*shuffle=*/0);
      k_inorm_prelu<<<Bn*64,256,0,stream>>>(convraw, 64, skip, 320, 192-64*il,
          dl[il].g, dl[il].bt, dl[il].a, (int)plane, 64);
    }
  };

  // ============================ encoder ============================
  k_conv1x1<<<(unsigned)CDIV((long)Bn*64*planeF,256),256,0,stream>>>(amp_in, pha_in, e_w1, e_b1, convraw, Bn,Tn,Ff);
  k_inorm_prelu<<<Bn*64,256,0,stream>>>(convraw, 64, skipF, 320, 256, e_g1, e_bt1, e_a1, (int)planeF, 64);
  run_dense(skipF, ed, Ff);
  {
    long Md = (long)Bn*Tn*F2;
    dim3 gg((unsigned)CDIV(Md,64),1);
    k_conv_wmma<<<gg,256,0,stream>>>(skipF, 320, 0, e_w2, e_b2, convraw,
        Bn, Tn, Ff, F2, 64, 64, /*KT=*/1, /*dil=*/1, /*stride=*/2, 0);
    k_inorm_prelu<<<Bn*64,256,0,stream>>>(convraw, 64, x_cur, 64, 0, e_g2, e_bt2, e_a2, (int)planeF2, 64);
  }

  // ========================= dual-path blocks ======================
  dim3 g64((unsigned)CDIV(M2,64),1);
  for (int kb=0;kb<4;kb++){
    const Blk& P = blk[kb];
    // --- time path (xLSTM) ---
    k_ln_rows<<<(unsigned)CDIV(M2,4),256,0,stream>>>(x_cur, xt_ln, xt_ln16, P.ln1_w, P.ln1_b, Bn,Tn,F2, 0);
    k_gemm_wmma<<<g64,256,0,stream>>>(xt_ln16,64, bw[kb].o, gates,256,0,   (int)M2,64,64, nullptr, nullptr,0, 0);
    k_gemm_wmma<<<g64,256,0,stream>>>(xt_ln16,64, bw[kb].i, gates,256,64,  (int)M2,64,64, nullptr, nullptr,0, 0);
    k_gemm_wmma<<<g64,256,0,stream>>>(xt_ln16,64, bw[kb].f, gates,256,128, (int)M2,64,64, nullptr, nullptr,0, 0);
    k_gemm_wmma<<<g64,256,0,stream>>>(xt_ln16,64, bw[kb].g, gates,256,192, (int)M2,64,64, nullptr, nullptr,0, 0);
    k_xlstm_scan<<<(unsigned)CDIV((long)Bn*F2*64,256),256,0,stream>>>(gates, hbuf16, Bn*F2, Tn);
    k_gemm_wmma<<<g64,256,0,stream>>>(hbuf16,64, bw[kb].out, xt_post,64,0, (int)M2,64,64, nullptr, xt_ln,64, 0);
    // --- freq path (mamba) ---
    k_ln_rows<<<(unsigned)CDIV(M2,4),256,0,stream>>>(xt_post, xf_ln, xf_ln16, P.ln2_w, P.ln2_b, Bn,Tn,F2, 1);
    float* xr = gates;
    k_gemm_wmma<<<dim3((unsigned)CDIV(M2,64),4),256,0,stream>>>(xf_ln16,64, bw[kb].inw, xr,256,0, (int)M2,256,64, nullptr,nullptr,0,0);
    k_dwconv_silu<<<(unsigned)CDIV(M2*128,256),256,0,stream>>>(xr, P.conv_w, P.conv_b, xma, xma16, F2, M2);
    k_gemm_wmma<<<dim3((unsigned)CDIV(M2,64),2),256,0,stream>>>(xma16,128, bw[kb].dtw, deltab,128,0, (int)M2,128,128, P.dt_b, nullptr,0, /*softplus*/1);
    k_gemm_wmma<<<g64,256,0,stream>>>(xma16,128, bw[kb].bw, bc,32,0,  (int)M2,16,128, nullptr,nullptr,0,0);
    k_gemm_wmma<<<g64,256,0,stream>>>(xma16,128, bw[kb].cw, bc,32,16, (int)M2,16,128, nullptr,nullptr,0,0);
    k_mamba_scan<<<(unsigned)CDIV((long)Bn*Tn*128,256),256,0,stream>>>(deltab, bc, xma, xr, P.A_log, P.Dp, ymid16, Bn*Tn, F2);
    k_gemm_wmma<<<g64,256,0,stream>>>(ymid16,128, bw[kb].ow, xf_post,64,0, (int)M2,64,128, nullptr, xf_ln,64, 0);
    k_untranspose_add<<<(unsigned)CDIV((long)Bn*64*planeF2,256),256,0,stream>>>(xf_post, x_cur, Bn,Tn,F2);
  }

  // =========================== decoders ============================
  auto run_decoder = [&](const DenseL* dl, const float* spw, const float* spb,
                         const float* g_, const float* bt_, const float* a_){
    k_copy_channels<<<(unsigned)CDIV((long)Bn*64*planeF2,256),256,0,stream>>>(x_cur, skipF2, (long)Bn*64*planeF2, planeF2);
    run_dense(skipF2, dl, F2);
    dim3 gg((unsigned)CDIV(M2,64), 2);
    k_conv_wmma<<<gg,256,0,stream>>>(skipF2, 320, 0, spw, spb, m202,
        Bn, Tn, F2, F2, 64, 128, /*KT=*/1, /*dil=*/1, /*stride=*/1, /*shuffle=*/1);
    k_inorm_prelu<<<Bn*64,256,0,stream>>>(m202, 64, m202, 64, 0, g_, bt_, a_, (int)plane202, 64);
  };

  run_decoder(md, m_sp_w, m_sp_b, m_g, m_bt, m_a);
  k_maskconv<<<(unsigned)CDIV(NT,256),256,0,stream>>>(m202, m_out_w, m_out_b, maskp, Bn, Tn, Ff, 202);

  run_decoder(pd, p_sp_w, p_sp_b, p_g, p_bt, p_a);
  k_phaseconv<<<(unsigned)CDIV(NT,256),256,0,stream>>>(m202, p_r_w, p_r_b, p_i_w, p_i_b, phap, Bn, Tn, Ff, 202);

  k_final<<<(unsigned)CDIV(NT,256),256,0,stream>>>(amp_in, maskp, phap, (float*)d_out, NT);
}
